// FFM_28252294873094
// MI455X (gfx1250) — matrix operand order repaired and checked by reading, not compiled, not run on hardware
//
#include <hip/hip_runtime.h>
#include <math.h>

typedef __attribute__((ext_vector_type(16))) _Float16 v16h;
typedef __attribute__((ext_vector_type(8)))  float    v8f;

#define F 6
#define LAT 16
#define SPB 16   // samples per block

// (field i, slot s) -> pair index and whether it is the "U" (i<j) or "W" side
__device__ __constant__ int kPair[30] = {
  0,1,2,3,4,   0,5,6,7,8,   1,5,9,10,11,
  2,6,9,12,13, 3,7,10,12,14, 4,8,11,13,14 };
__device__ __constant__ int kIsU[30] = {
  1,1,1,1,1,  0,1,1,1,1,  0,0,1,1,1,
  0,0,0,1,1,  0,0,0,0,1,  0,0,0,0,0 };

__global__ __launch_bounds__(128) void ffm_wmma_kernel(
    const int* __restrict__ x0, const int* __restrict__ x1, const int* __restrict__ x2,
    const int* __restrict__ x3, const int* __restrict__ x4, const int* __restrict__ x5,
    const float* __restrict__ E0, const float* __restrict__ E1, const float* __restrict__ E2,
    const float* __restrict__ E3, const float* __restrict__ E4, const float* __restrict__ E5,
    const float* __restrict__ L0, const float* __restrict__ L1, const float* __restrict__ L2,
    const float* __restrict__ L3, const float* __restrict__ L4, const float* __restrict__ L5,
    const float* __restrict__ Wd, const float* __restrict__ bd,
    float* __restrict__ out)
{
    // U/W staged as f16 so phase-2 fragment loads are cheap; pair 15 is zero pad.
    __shared__ _Float16 Ubuf[SPB][16][LAT];   // 8 KB
    __shared__ _Float16 Wbuf[SPB][16][LAT];   // 8 KB
    __shared__ float    linbuf[SPB];
    __shared__ float    Dtile[16][16];        // diagonal extraction

    const int tid  = threadIdx.x;
    const int base = blockIdx.x * SPB;        // first sample of this block

    const int* xs[F]      = { x0, x1, x2, x3, x4, x5 };
    const float* Es[F]    = { E0, E1, E2, E3, E4, E5 };
    const float* Ls[F]    = { L0, L1, L2, L3, L4, L5 };
    const int   dims[F]   = { 1000000, 500000, 500000, 100000, 10000, 1000 };
    const int   seq[F]    = { 1, 1, 50, 20, 1, 1 };

    // ---- zero the pad pair (index 15) ----
    for (int k = tid; k < SPB * LAT; k += 128) {
        int b = k >> 4, l = k & 15;
        Ubuf[b][15][l] = (_Float16)0.0f;
        Wbuf[b][15][l] = (_Float16)0.0f;
    }

    // ---- phase 1: gather + mean-pool all 30 (field,slot) embeddings ----
    // 8 groups of 16 lanes; lane = latent dim -> coalesced 64B row reads.
    const int g = tid >> 4;        // group 0..7
    const int l = tid & 15;        // latent lane
    #pragma unroll
    for (int c0 = 0; c0 < 30; ++c0) {
        const int i = c0 / 5;                 // field (compile-time after unroll)
        const int s = c0 % 5;                 // slot
        const int n = seq[i];
        const int p = kPair[c0];
        const float* Ep = Es[i] + (size_t)s * (size_t)dims[i] * LAT;
        for (int b = g; b < SPB; b += 8) {    // 2 samples per group per combo
            const int* xp = xs[i] + (size_t)(base + b) * n;
            float acc = 0.0f;
            for (int t = 0; t < n; ++t) {
                int idx = xp[t];
                acc += Ep[(size_t)idx * LAT + l];
            }
            _Float16 h = (_Float16)(acc * (1.0f / (float)n));
            if (kIsU[c0]) Ubuf[b][p][l] = h;
            else          Wbuf[b][p][l] = h;
        }
    }

    // ---- linear term (tiny): lanes 0..15, one sample each ----
    if (tid < SPB) {
        const int b = tid;
        float lin = bd[0];
        #pragma unroll
        for (int i = 0; i < F; ++i) {
            const int* xp = xs[i] + (size_t)(base + b) * seq[i];
            float f = 0.0f;
            for (int t = 0; t < seq[i]; ++t) f += Ls[i][xp[t]];
            lin += (f * (1.0f / (float)seq[i])) * Wd[i];
        }
        linbuf[b] = fmaxf(lin, 0.0f);         // relu
    }
    __syncthreads();

    // ---- phase 2: wave 0 reduces all 15 pair dot products for 16 samples ----
    // A row m = sample m's U (240 vals + pad), B col n = sample n's W.
    // D[m][m] = fm[m]; 8 accumulating v_wmma_f32_16x16x32_f16 cover K=256.
    if (tid < 32) {
        const int lane = tid;
        const int row  = lane & 15;                 // A row / B col = sample id
        const int hiA  = (lane >> 4) * 8;           // A: +8 K-offset for lanes 16..31
        const int hiB  = (lane >> 4) * 16;          // unused; B also takes hiA (+8) K-offset
        const _Float16* Uf = &Ubuf[row][0][0];      // 256 contiguous halfs
        const _Float16* Wf = &Wbuf[row][0][0];
        v8f acc = {};
        #pragma unroll
        for (int c = 0; c < 8; ++c) {
            const int k0 = 32 * c;
            v16h a, bm;
            #pragma unroll
            for (int e = 0; e < 16; ++e) {
                const int ka = k0 + e + ((e >= 8) ? 8 : 0) + hiA; // 16-bit A 16x32 layout
                const int kb = k0 + e + ((e >= 8) ? 8 : 0) + hiA; // 16-bit B 32x16 layout
                a[e]  = Uf[ka];
                bm[e] = Wf[kb];
            }
            acc = __builtin_amdgcn_wmma_f32_16x16x32_f16(
                      /*neg_a=*/false, a, /*neg_b=*/false, bm,
                      /*c_mod=*/(short)0, acc, /*reuse_a=*/false, /*reuse_b=*/false);
        }
        // scatter D to LDS tile (C/D layout: vgpr r -> rows r / r+8), read diagonal
        const int col    = lane & 15;
        const int rowOff = (lane >> 4) * 8;
        #pragma unroll
        for (int r = 0; r < 8; ++r)
            Dtile[r + rowOff][col] = acc[r];
        // same-wave DS ops are in-order; compiler inserts s_wait_dscnt
        if (lane < SPB) {
            float fm = Dtile[lane][lane];
            float v  = linbuf[lane] + fm;
            out[base + lane] = 1.0f / (1.0f + expf(-v));
        }
    }
}

extern "C" void kernel_launch(void* const* d_in, const int* in_sizes, int n_in,
                              void* d_out, int out_size, void* d_ws, size_t ws_size,
                              hipStream_t stream) {
    const int* x0 = (const int*)d_in[0];
    const int* x1 = (const int*)d_in[1];
    const int* x2 = (const int*)d_in[2];
    const int* x3 = (const int*)d_in[3];
    const int* x4 = (const int*)d_in[4];
    const int* x5 = (const int*)d_in[5];
    const float* E0 = (const float*)d_in[6];
    const float* E1 = (const float*)d_in[7];
    const float* E2 = (const float*)d_in[8];
    const float* E3 = (const float*)d_in[9];
    const float* E4 = (const float*)d_in[10];
    const float* E5 = (const float*)d_in[11];
    const float* L0 = (const float*)d_in[12];
    const float* L1 = (const float*)d_in[13];
    const float* L2 = (const float*)d_in[14];
    const float* L3 = (const float*)d_in[15];
    const float* L4 = (const float*)d_in[16];
    const float* L5 = (const float*)d_in[17];
    const float* Wd = (const float*)d_in[18];
    const float* bd = (const float*)d_in[19];
    float* out = (float*)d_out;

    const int batch  = in_sizes[0];        // x0 is (B,1)
    const int blocks = batch / SPB;        // 16384 / 16 = 1024
    hipLaunchKernelGGL(ffm_wmma_kernel, dim3(blocks), dim3(128), 0, stream,
                       x0, x1, x2, x3, x4, x5,
                       E0, E1, E2, E3, E4, E5,
                       L0, L1, L2, L3, L4, L5,
                       Wd, bd, out);
}
